// U_Paddle_68874095559201
// MI455X (gfx1250) — matrix operand order repaired and checked by reading, not compiled, not run on hardware
//
#include <hip/hip_runtime.h>
#include <stdint.h>

// ---------------- problem constants ----------------
#define NW      5      // classes K
#define SS      25     // support set size
#define QQ      75     // query set size
#define DD      640    // feature dim
#define QP      96     // Q padded to multiple of 32 (WMMA K-chunk)
#define KP      16     // classes padded to 16 (WMMA M)
#define LAYERS  20

// ---------------- LDS layout (bytes, all 16B aligned) ----------------
// padded row lengths (halves) chosen for bank-conflict-free b128 loads
#define ZQROW 648   // Zq16 row:  1296B -> lane rows land on distinct banks
#define ZTROW 104   // Zqt row:    208B
#define WROW  648   // W16 row
#define UTROW 104   // Ut16 row

#define OFF_ZQ   0
#define SZ_ZQ    (QP * ZQROW * 2)          // 124416
#define OFF_ZT   (OFF_ZQ + SZ_ZQ)
#define SZ_ZT    (DD * ZTROW * 2)          // 133120
#define OFF_W    (OFF_ZT + SZ_ZT)
#define SZ_W     (KP * WROW * 2)           // 20736
#define OFF_UT   (OFF_W + SZ_W)
#define SZ_UT    (KP * UTROW * 2)          // 3328
#define OFF_LG   (OFF_UT + SZ_UT)
#define SZ_LG    (KP * QP * 4)             // 6144
#define OFF_SN   (OFF_LG + SZ_LG)
#define SZ_SN    (NW * DD * 4)             // 12800
#define OFF_MISC (OFF_SN + SZ_SN)
#define SZ_MISC  1024                      // counts/V/bias/sqpart/gamma
#define OFF_YS   (OFF_MISC + SZ_MISC)
#define SZ_YS    128
#define SMEM_BYTES (OFF_YS + SZ_YS)        // 301696 < 320KB

static_assert(SMEM_BYTES <= 320 * 1024, "LDS budget exceeded");

typedef __attribute__((ext_vector_type(16))) _Float16 v16h;
typedef __attribute__((ext_vector_type(8)))  _Float16 v8h;
typedef __attribute__((ext_vector_type(8)))  float    v8f;

// Assemble a 16-half WMMA operand from two contiguous 16B LDS chunks
// (chunk0 -> vector halves 0..7, chunk1 -> halves 8..15).
__device__ __forceinline__ v16h ld2(const _Float16* p0, const _Float16* p1) {
    union { v16h v; v8h h[2]; } u;
    u.h[0] = *(const v8h*)p0;
    u.h[1] = *(const v8h*)p1;
    return u.v;
}

__global__ __launch_bounds__(256) void
paddle_fewshot_kernel(const float* __restrict__ Zs,   // [B,25,640]
                      const int*   __restrict__ Ys,   // [B,25]
                      const float* __restrict__ Zq,   // [B,75,640]
                      const float* __restrict__ gam,  // [1]
                      float*       __restrict__ out)  // [B,75,5]
{
    extern __shared__ char smem[];
    _Float16* zq16  = (_Float16*)(smem + OFF_ZQ);   // [96][648]  row-major f16
    _Float16* zqt   = (_Float16*)(smem + OFF_ZT);   // [640][104] d-major f16
    _Float16* w16   = (_Float16*)(smem + OFF_W);    // [16][648]
    _Float16* ut16  = (_Float16*)(smem + OFF_UT);   // [16][104]  U transposed
    float*    lg    = (float*)(smem + OFF_LG);      // [16][96]   logits^T
    float*    supn  = (float*)(smem + OFF_SN);      // [5][640]
    float*    cnt   = (float*)(smem + OFF_MISC);    // [16]
    float*    Vv    = cnt + 16;                     // [16]
    float*    bias  = cnt + 32;                     // [16]
    float*    den   = cnt + 48;                     // [16]
    float*    sqpt  = cnt + 64;                     // [160] partial ||w||^2
    float*    gls   = cnt + 232;                    // [1]
    int*      Ysh   = (int*)(smem + OFF_YS);        // [25]

    const int tid  = threadIdx.x;
    const int b    = blockIdx.x;
    const int lane = tid & 31;
    const int wid  = tid >> 5;
    const int m16  = lane & 15;     // WMMA row (A) / col (B) within tile
    const int hh   = lane >> 4;     // half-wave selector

    // ---- zero entire LDS (also zero-pads Zq rows 75..95, Ut, W rows 5..15) ----
    for (int i = tid; i < SMEM_BYTES / 4; i += 256) ((uint32_t*)smem)[i] = 0u;
    __syncthreads();

    if (tid < SS)  Ysh[tid] = Ys[(size_t)b * SS + tid];
    if (tid == 0)  gls[0] = gam[0];
    __syncthreads();
    const float g = gls[0];

    // ---- counts + sup_num = one_hot(Y)^T @ Z_support ----
    if (tid < NW) {
        int c = 0;
        for (int s = 0; s < SS; ++s) c += (Ysh[s] == tid);
        cnt[tid] = (float)c;
    }
    for (int col = tid; col < DD; col += 256) {
        float a0 = 0, a1 = 0, a2 = 0, a3 = 0, a4 = 0;
        const float* zp = Zs + (size_t)b * SS * DD + col;
        for (int s = 0; s < SS; ++s) {
            int y = Ysh[s];
            float z = zp[(size_t)s * DD];
            a0 += (y == 0) ? z : 0.f;  a1 += (y == 1) ? z : 0.f;
            a2 += (y == 2) ? z : 0.f;  a3 += (y == 3) ? z : 0.f;
            a4 += (y == 4) ? z : 0.f;
        }
        supn[0 * DD + col] = a0; supn[1 * DD + col] = a1; supn[2 * DD + col] = a2;
        supn[3 * DD + col] = a3; supn[4 * DD + col] = a4;
    }
    __syncthreads();

    // ---- W0 = sup_num / counts  (f16, padded rows stay 0); V0 = 1 ----
    for (int i = tid; i < KP * DD; i += 256) {
        int m = i / DD, d = i - m * DD;
        float w = (m < NW) ? supn[m * DD + d] / cnt[m] : 0.f;
        w16[m * WROW + d] = (_Float16)w;
    }
    if (tid < KP) Vv[tid] = 1.f;
    // ---- stage Z_query into LDS twice (row-major + d-major) as f16 ----
    {
        const float* zp = Zq + (size_t)b * QQ * DD;
        for (int i = tid; i < QQ * DD; i += 256) {
            int q = i / DD, d = i - q * DD;
            _Float16 h = (_Float16)zp[i];
            zq16[q * ZQROW + d] = h;
            zqt[d * ZTROW + q]  = h;
        }
    }
    __syncthreads();

    // ---- ||W0||^2 partials -> bias0 = g*1 - 0.5*||w||^2 ----
    if (tid < NW * 32) {
        int c = tid >> 5, j = tid & 31;
        float s = 0;
        const _Float16* w = w16 + c * WROW;
        for (int d = j; d < DD; d += 32) { float x = (float)w[d]; s += x * x; }
        sqpt[tid] = s;
    }
    __syncthreads();
    if (tid < NW) {
        float s = 0;
        #pragma unroll
        for (int j = 0; j < 32; ++j) s += sqpt[tid * 32 + j];
        bias[tid] = g * Vv[tid] - 0.5f * s;
    }
    __syncthreads();

    // ================= layer loop =================
    for (int layer = 0; layer < LAYERS; ++layer) {
        // ---- GEMM1: logits^T[16c x 96q] = W[16x640] @ Zq^T ----
        if (wid < QP / 16) {                      // 6 q-tiles over waves 0..5
            int q0 = wid << 4;
            v8f acc = {};
            const _Float16* ar = w16 + m16 * WROW;               // A row = class
            const _Float16* br = zq16 + (q0 + m16) * ZQROW;      // B col = query
            for (int d0 = 0; d0 < DD; d0 += 32) {
                v16h a  = ld2(ar + d0 + 8 * hh, ar + d0 + 16 + 8 * hh);
                v16h bm = ld2(br + d0 + 8 * hh, br + d0 + 16 + 8 * hh);
                acc = __builtin_amdgcn_wmma_f32_16x16x32_f16(
                          false, a, false, bm, (short)0, acc, false, false);
            }
            #pragma unroll
            for (int r = 0; r < 8; ++r)
                lg[(r + 8 * hh) * QP + q0 + m16] = acc[r];
        }
        __syncthreads();

        // ---- softmax over classes (the -0.5*||zq||^2 term cancels) ----
        if (tid < QQ) {
            int q = tid;
            float l0 = lg[0 * QP + q] + bias[0];
            float l1 = lg[1 * QP + q] + bias[1];
            float l2 = lg[2 * QP + q] + bias[2];
            float l3 = lg[3 * QP + q] + bias[3];
            float l4 = lg[4 * QP + q] + bias[4];
            float mx = fmaxf(fmaxf(fmaxf(l0, l1), fmaxf(l2, l3)), l4);
            float e0 = __expf(l0 - mx), e1 = __expf(l1 - mx), e2 = __expf(l2 - mx);
            float e3 = __expf(l3 - mx), e4 = __expf(l4 - mx);
            float inv = 1.f / (e0 + e1 + e2 + e3 + e4);
            float u0 = e0 * inv, u1 = e1 * inv, u2 = e2 * inv, u3 = e3 * inv, u4 = e4 * inv;
            ut16[0 * UTROW + q] = (_Float16)u0;
            ut16[1 * UTROW + q] = (_Float16)u1;
            ut16[2 * UTROW + q] = (_Float16)u2;
            ut16[3 * UTROW + q] = (_Float16)u3;
            ut16[4 * UTROW + q] = (_Float16)u4;
            if (layer == LAYERS - 1) {
                float* o = out + ((size_t)b * QQ + q) * NW;
                o[0] = u0; o[1] = u1; o[2] = u2; o[3] = u3; o[4] = u4;
            }
        }
        __syncthreads();
        if (layer == LAYERS - 1) break;

        // ---- V, den (deterministic serial reduction over 75 f16 values) ----
        if (tid < NW) {
            float s = 0;
            const _Float16* u = ut16 + tid * UTROW;
            for (int q = 0; q < QQ; ++q) s += (float)u[q];
            Vv[tid]  = __logf(s * (1.0f / QQ) + 1e-6f) + 1.0f;
            den[tid] = s + cnt[tid];
        }
        __syncthreads();

        // ---- GEMM2: num[16c x 640d] = U^T @ Zq ; W = (num+sup_num)/den ----
        for (int t = wid; t < DD / 16; t += 8) {  // 40 d-tiles over 8 waves
            int d0 = t << 4;
            v8f acc = {};
            const _Float16* ar = ut16 + m16 * UTROW;             // A row = class
            const _Float16* br = zqt + (d0 + m16) * ZTROW;       // B col = feature
            for (int q0 = 0; q0 < QP; q0 += 32) {
                v16h a  = ld2(ar + q0 + 8 * hh, ar + q0 + 16 + 8 * hh);
                v16h bm = ld2(br + q0 + 8 * hh, br + q0 + 16 + 8 * hh);
                acc = __builtin_amdgcn_wmma_f32_16x16x32_f16(
                          false, a, false, bm, (short)0, acc, false, false);
            }
            #pragma unroll
            for (int r = 0; r < 8; ++r) {
                int c = r + 8 * hh;
                int d = d0 + m16;
                float w = 0.f;
                if (c < NW) w = (acc[r] + supn[c * DD + d]) / den[c];
                w16[c * WROW + d] = (_Float16)w;
            }
        }
        __syncthreads();

        // ---- ||W||^2 partials, then bias = g*V - 0.5*||w||^2 ----
        if (tid < NW * 32) {
            int c = tid >> 5, j = tid & 31;
            float s = 0;
            const _Float16* w = w16 + c * WROW;
            for (int d = j; d < DD; d += 32) { float x = (float)w[d]; s += x * x; }
            sqpt[tid] = s;
        }
        __syncthreads();
        if (tid < NW) {
            float s = 0;
            #pragma unroll
            for (int j = 0; j < 32; ++j) s += sqpt[tid * 32 + j];
            bias[tid] = g * Vv[tid] - 0.5f * s;
        }
        __syncthreads();
    }
}

extern "C" void kernel_launch(void* const* d_in, const int* in_sizes, int n_in,
                              void* d_out, int out_size, void* d_ws, size_t ws_size,
                              hipStream_t stream) {
    (void)n_in; (void)out_size; (void)d_ws; (void)ws_size;
    const float* Zs = (const float*)d_in[0];   // Z_support f32
    const int*   Ys = (const int*)d_in[1];     // Y_support i32
    const float* Zq = (const float*)d_in[2];   // Z_query   f32
    const float* g  = (const float*)d_in[3];   // gamma     f32 scalar
    float* out = (float*)d_out;

    const int B = in_sizes[1] / SS;            // tasks

    (void)hipFuncSetAttribute((const void*)paddle_fewshot_kernel,
                              hipFuncAttributeMaxDynamicSharedMemorySize,
                              SMEM_BYTES);
    paddle_fewshot_kernel<<<B, 256, SMEM_BYTES, stream>>>(Zs, Ys, Zq, g, out);
}
